// LLaMA3Layer_85770496901174
// MI455X (gfx1250) — compile-verified, matrix operand order checked by reading, not run on hardware
//
#include <hip/hip_runtime.h>
#include <hip/hip_bf16.h>

#define D_MODEL  2048
#define N_HEADS  16
#define HEAD_DIM 128
#define D_FF     8192

typedef __attribute__((ext_vector_type(16))) __bf16       v16bf;
typedef __attribute__((ext_vector_type(8)))  float        v8f;
typedef __attribute__((ext_vector_type(4)))  int          v4i;
typedef __attribute__((ext_vector_type(8)))  int          v8i;
typedef __attribute__((ext_vector_type(4)))  unsigned int v4u;

#ifndef __has_builtin
#define __has_builtin(x) 0
#endif

// ---- CDNA5 async global->LDS copy (ASYNCcnt-tracked) ----------------------
#if __has_builtin(__builtin_amdgcn_global_load_async_to_lds_b128)
#define HAVE_ASYNC_LDS 1
__device__ __forceinline__ void async_copy16(const void* g, void* l) {
  typedef __attribute__((address_space(1))) v4i* gp_t;
  typedef __attribute__((address_space(3))) v4i* lp_t;
  __builtin_amdgcn_global_load_async_to_lds_b128((gp_t)(void*)(g), (lp_t)(l), 0, 0);
}
__device__ __forceinline__ void wait_async0() {
#if __has_builtin(__builtin_amdgcn_s_wait_asynccnt)
  __builtin_amdgcn_s_wait_asynccnt(0);
#else
  asm volatile("s_wait_asynccnt 0x0" ::: "memory");
#endif
}
#else
#define HAVE_ASYNC_LDS 0
__device__ __forceinline__ void wait_async0() {}
#endif

// ---- CDNA5 Tensor Data Mover (TENSORcnt-tracked) --------------------------
// This toolchain exposes the 6-arg form:
//   (uint32x4 g0, int32x8 g1, int32x4 g2, int32x4 g3, int32x8, i32 cpol)
// D# per CDNA5 ISA: g0 = {count, lds_addr, global_addr, type=2};
// g1 = {mask/data_size/pad, tensor dims, tile dims, dim0 stride}.
#if __has_builtin(__builtin_amdgcn_tensor_load_to_lds) && __has_builtin(__builtin_amdgcn_s_wait_tensorcnt)
#define HAVE_TDM 1
// Load a 32-row x 128-elem bf16 tile (row stride 2048 elems in memory) into
// LDS with 4 DWORDs of pad every 64 DWORDs (-> 136-elem LDS rows).
__device__ __forceinline__ void tdm_load_k_tile(const void* gsrc, void* ldst) {
  const unsigned long long ga = (unsigned long long)(size_t)gsrc;
  const unsigned lds = (unsigned)(size_t)ldst;
  v4u g0;
  g0[0] = 1u;                                           // count=1 (valid), user
  g0[1] = lds;                                          // lds_addr
  g0[2] = (unsigned)ga;                                 // global_addr[31:0]
  g0[3] = (unsigned)((ga >> 32) & 0x01FFFFFFu) | (2u << 30);  // ga[56:32], type=2
  v8i g1;
  g1[0] = (int)((1u << 16)        // data_size = 1 (2 bytes)
              | (1u << 20)        // pad_enable
              | (5u << 22)        // pad_interval: 64 DWORDs (one 256B row)
              | (3u << 25));      // pad_amount: 4 DWORDs (16B)
  g1[1] = (int)(128u << 16);      // tensor_dim0 = 128  (bits 79:48)
  g1[2] = (int)(32u << 16);       // tensor_dim1 = 32   (bits 111:80)
  g1[3] = (int)(128u << 16);      // tile_dim0 = 128    (bits 127:112)
  g1[4] = 32;                     // tile_dim1 = 32, tile_dim2 = 0
  g1[5] = 2048;                   // tensor_dim0_stride = 2048 elems (bits 207:160)
  g1[6] = 0;
  g1[7] = 0;
  const v4i gz4 = {0, 0, 0, 0};                      // 2D tensor: groups 2/3 unused
  const v8i gz8 = {0, 0, 0, 0, 0, 0, 0, 0};
  __builtin_amdgcn_tensor_load_to_lds(g0, g1, gz4, gz4, gz8, 0);
}
__device__ __forceinline__ void wait_tensor0() {
  __builtin_amdgcn_s_wait_tensorcnt(0);
}
#else
#define HAVE_TDM 0
__device__ __forceinline__ void wait_tensor0() {}
#endif

__device__ __forceinline__ __bf16 f2bf(float f) {
  unsigned u = __builtin_bit_cast(unsigned, f);
  u += 0x7FFFu + ((u >> 16) & 1u);              // round-to-nearest-even
  unsigned short s = (unsigned short)(u >> 16);
  return __builtin_bit_cast(__bf16, s);
}

__device__ __forceinline__ v8f vzero() {
  v8f z = {0.f, 0.f, 0.f, 0.f, 0.f, 0.f, 0.f, 0.f};
  return z;
}

// A/B fragment load for V_WMMA_F32_16X16X32_BF16 from K-contiguous memory.
// Lane L (L<16) holds row L, K = {0..7, 16..23}; lane L+16 holds row L,
// K = {8..15, 24..31}. Caller passes p = row_base + (lane>>4)*8.
__device__ __forceinline__ v16bf ldfrag(const __bf16* p) {
  v16bf f;
  uint4* d = (uint4*)&f;
  d[0] = *(const uint4*)(p);
  d[1] = *(const uint4*)(p + 16);
  return f;
}

__device__ __forceinline__ v8f wmma_bf16(v16bf a, v16bf b, v8f c) {
  return __builtin_amdgcn_wmma_f32_16x16x32_bf16(
      false, a, false, b, (short)0, c, false, false);
}

// ---------------------------------------------------------------------------
// RMSNorm: fp32 row -> bf16 row, one block per token row.
// ---------------------------------------------------------------------------
__global__ __launch_bounds__(256)
void rmsnorm_kernel(const float* __restrict__ x, const float* __restrict__ w,
                    __bf16* __restrict__ out, int D) {
  const int row = blockIdx.x;
  const int tid = threadIdx.x;
  const float* xr = x + (size_t)row * D;
  float ss = 0.f;
  for (int i = tid; i < D; i += 256) { float v = xr[i]; ss += v * v; }
  __shared__ float red[256];
  red[tid] = ss;
  __syncthreads();
  for (int s = 128; s > 0; s >>= 1) {
    if (tid < s) red[tid] += red[tid + s];
    __syncthreads();
  }
  const float inv = rsqrtf(red[0] / (float)D + 1e-8f);
  __bf16* orow = out + (size_t)row * D;
  for (int i = tid; i < D; i += 256) orow[i] = f2bf(xr[i] * inv * w[i]);
}

// ---------------------------------------------------------------------------
// Tiled WMMA GEMM:  out[M,N] = A[M,K](bf16) @ Bw[N,K](fp32)^T + bias
//   ROTARY : add rotary[col % 128] (q projection)
//   RESID  : fp32 output with residual add; else bf16 output
//   SWIGLU : Bw has 2N rows; out = (A@B1^T+b1) * silu(A@B2^T+b2), bf16
// Block tile 128x64, K-step 32, 8 waves (4x2), 32x32 per wave.
// Double-buffered LDS (80B-padded rows), ASYNCcnt staging overlapped with
// WMMA, global_prefetch on the tile after next.
// ---------------------------------------------------------------------------
template <bool ROTARY, bool RESID, bool SWIGLU>
__global__ __launch_bounds__(256)
void gemm_wmma_kernel(const __bf16* __restrict__ A, const float* __restrict__ Bw,
                      const float* __restrict__ bias, const float* __restrict__ resid,
                      const float* __restrict__ rot, void* __restrict__ outp,
                      int M, int N, int K) {
  __shared__ __bf16 As[2][128 * 40];
  __shared__ __bf16 Bs[2][64 * 40];
  __shared__ __bf16 Bs2[2][64 * 40];

  const int tid  = threadIdx.x;
  const int wave = tid >> 5;
  const int lane = tid & 31;
  const int wm   = wave >> 1;   // 0..3
  const int wn   = wave & 1;    // 0..1
  const int m0   = blockIdx.x * 128;
  const int n0   = blockIdx.y * 64;

  auto stageA = [&](int kk, int buf) {
    for (int i = tid; i < 512; i += 256) {
      const int r = i >> 2, c = (i & 3) * 8;
      const __bf16* g = &A[(size_t)(m0 + r) * K + kk + c];
      __bf16* l = &As[buf][r * 40 + c];
#if HAVE_ASYNC_LDS
      async_copy16(g, l);
#else
      *(uint4*)l = *(const uint4*)g;
#endif
    }
  };
  auto stageB = [&](int kk, int buf) {
    const int r = tid >> 2, c = (tid & 3) * 8;
    {
      const float* src = &Bw[(size_t)(n0 + r) * K + kk + c];
      alignas(16) __bf16 tmp[8];
#pragma unroll
      for (int j = 0; j < 8; j++) tmp[j] = f2bf(src[j]);
      *(uint4*)&Bs[buf][r * 40 + c] = *(const uint4*)tmp;
    }
    if (SWIGLU) {
      const float* src2 = &Bw[(size_t)(n0 + r + N) * K + kk + c];
      alignas(16) __bf16 tmp2[8];
#pragma unroll
      for (int j = 0; j < 8; j++) tmp2[j] = f2bf(src2[j]);
      *(uint4*)&Bs2[buf][r * 40 + c] = *(const uint4*)tmp2;
    }
    if (kk + 32 < K) {  // prefetch the tile after next (global_prefetch_b8)
      __builtin_prefetch(&Bw[(size_t)(n0 + r) * K + kk + 32 + c], 0, 3);
      if (SWIGLU)
        __builtin_prefetch(&Bw[(size_t)(n0 + r + N) * K + kk + 32 + c], 0, 3);
    }
  };

  v8f c00 = vzero(), c01 = vzero(), c10 = vzero(), c11 = vzero();
  v8f d00 = vzero(), d01 = vzero(), d10 = vzero(), d11 = vzero();

  const int arow = (wm * 32 + (lane & 15)) * 40 + ((lane >> 4) * 8);
  const int brow = (wn * 32 + (lane & 15)) * 40 + ((lane >> 4) * 8);

  int cur = 0;
  stageA(0, 0);
  stageB(0, 0);
  wait_async0();
  __syncthreads();

  for (int k0 = 0; k0 < K; k0 += 32) {
    // 1) fragment register loads from the live buffer (ds ops first)
    const v16bf a0 = ldfrag(&As[cur][arow]);
    const v16bf a1 = ldfrag(&As[cur][arow + 16 * 40]);
    const v16bf b0 = ldfrag(&Bs[cur][brow]);
    const v16bf b1 = ldfrag(&Bs[cur][brow + 16 * 40]);
    v16bf e0 = b0, e1 = b1;
    if (SWIGLU) {
      e0 = ldfrag(&Bs2[cur][brow]);
      e1 = ldfrag(&Bs2[cur][brow + 16 * 40]);
    }
    // 2) kick off staging of the next tile into the other buffer
    if (k0 + 32 < K) {
      stageA(k0 + 32, cur ^ 1);
      stageB(k0 + 32, cur ^ 1);
    }
    // 3) matrix math overlaps the in-flight staging
    c00 = wmma_bf16(a0, b0, c00);
    c01 = wmma_bf16(a0, b1, c01);
    c10 = wmma_bf16(a1, b0, c10);
    c11 = wmma_bf16(a1, b1, c11);
    if (SWIGLU) {
      d00 = wmma_bf16(a0, e0, d00);
      d01 = wmma_bf16(a0, e1, d01);
      d10 = wmma_bf16(a1, e0, d10);
      d11 = wmma_bf16(a1, e1, d11);
    }
    wait_async0();
    __syncthreads();
    cur ^= 1;
  }

  // Epilogue. C layout: VGPR r -> row (r + 8*(lane>>4)), col = lane&15.
  const int rb = (lane >> 4) * 8;
  const int cl = lane & 15;
  v8f cs[2][2]  = {{c00, c01}, {c10, c11}};
  v8f dsv[2][2] = {{d00, d01}, {d10, d11}};
#pragma unroll
  for (int mt = 0; mt < 2; mt++) {
#pragma unroll
    for (int nt = 0; nt < 2; nt++) {
#pragma unroll
      for (int r = 0; r < 8; r++) {
        const int row = m0 + wm * 32 + mt * 16 + rb + r;
        const int col = n0 + wn * 32 + nt * 16 + cl;
        float val = cs[mt][nt][r] + bias[col];
        if (SWIGLU) {
          const float x2 = dsv[mt][nt][r] + bias[col + N];
          val = val * (x2 / (1.f + __expf(-x2)));
        }
        if (ROTARY) val += rot[col & (HEAD_DIM - 1)];
        const size_t oi = (size_t)row * N + col;
        if (RESID) ((float*)outp)[oi]  = val + resid[oi];
        else       ((__bf16*)outp)[oi] = f2bf(val);
      }
    }
  }
}

// ---------------------------------------------------------------------------
// Flash attention (no mask): one block per (b,h, 64 q-rows); 4 waves,
// 16 q-rows per wave, streaming double-buffered 32-kv tiles with online
// softmax. K tile staged via the Tensor Data Mover (one descriptor per
// block, TENSORcnt) with LDS row padding; V tile staged transposed by the
// block. Fragments are preloaded in batches so WMMAs issue back-to-back.
// q,k,v,o are bf16 [B,S,H,HEAD_DIM] (= [M, D_MODEL]).
// ---------------------------------------------------------------------------
__global__ __launch_bounds__(128)
void flash_attn_kernel(const __bf16* __restrict__ q, const __bf16* __restrict__ k,
                       const __bf16* __restrict__ v, __bf16* __restrict__ o,
                       int S) {
  __shared__ __bf16 Ks[2][32 * 136];   // kv-major, 272B rows (TDM pad 16B/row)
  __shared__ __bf16 Vt[2][128 * 40];   // d-major (transposed), 80B-padded rows
  __shared__ __bf16 Pl[4][16 * 40];    // per-wave probability tile

  const int bh   = blockIdx.x;
  const int b    = bh / N_HEADS;
  const int h    = bh % N_HEADS;
  const int tid  = threadIdx.x;
  const int wave = tid >> 5;
  const int lane = tid & 31;
  const int q0   = blockIdx.y * 64 + wave * 16;
  const size_t rs = (size_t)D_MODEL;
  const __bf16* qb = q + ((size_t)b * S) * rs + h * HEAD_DIM;
  const __bf16* kb = k + ((size_t)b * S) * rs + h * HEAD_DIM;
  const __bf16* vb = v + ((size_t)b * S) * rs + h * HEAD_DIM;

  auto stage = [&](int kv0, int buf) {
#if HAVE_TDM
    if (wave == 0)  // one DMA descriptor per block (TDM ignores EXEC)
      tdm_load_k_tile(&kb[(size_t)kv0 * rs], &Ks[buf][0]);
#endif
    for (int i = tid; i < 512; i += 128) {
      const int kr = i >> 4;
      const int dc = (i & 15) * 8;
#if !HAVE_TDM
      const __bf16* gk = &kb[(size_t)(kv0 + kr) * rs + dc];
      __bf16* lk = &Ks[buf][kr * 136 + dc];
#if HAVE_ASYNC_LDS
      async_copy16(gk, lk);
#else
      *(uint4*)lk = *(const uint4*)gk;
#endif
#endif
      union { uint4 u; __bf16 e[8]; } t;
      t.u = *(const uint4*)&vb[(size_t)(kv0 + kr) * rs + dc];
#pragma unroll
      for (int j = 0; j < 8; j++) Vt[buf][(dc + j) * 40 + kr] = t.e[j];
    }
  };

  // Q fragments for the wave's 16 rows x 128 dims (4 K-chunks of 32)
  v16bf qf[4];
  {
    const __bf16* p = qb + (size_t)(q0 + (lane & 15)) * rs + (lane >> 4) * 8;
#pragma unroll
    for (int kc = 0; kc < 4; kc++) qf[kc] = ldfrag(p + kc * 32);
  }

  v8f oacc[8];
#pragma unroll
  for (int n = 0; n < 8; n++) oacc[n] = vzero();
  float mrow[8], lrow[8];
#pragma unroll
  for (int r = 0; r < 8; r++) { mrow[r] = -3.0e38f; lrow[r] = 0.f; }

  const float scale = 0.08838834764831845f;  // 1/sqrt(128)
  const int rb = (lane >> 4) * 8;
  const int cl = lane & 15;
  const int kbase = (lane & 15) * 136 + (lane >> 4) * 8;

  int buf = 0;
  stage(0, 0);
  for (int kv0 = 0; kv0 < S; kv0 += 32) {
    wait_async0();
    wait_tensor0();
    __syncthreads();                    // tile for this iteration is visible
    if (kv0 + 32 < S) stage(kv0 + 32, buf ^ 1);   // overlaps score compute

    // Preload all 8 K fragments, then run the 8 score WMMAs back-to-back.
    v16bf kf0[4], kf1[4];
#pragma unroll
    for (int kc = 0; kc < 4; kc++) kf0[kc] = ldfrag(&Ks[buf][kbase + kc * 32]);
#pragma unroll
    for (int kc = 0; kc < 4; kc++) kf1[kc] = ldfrag(&Ks[buf][kbase + 16 * 136 + kc * 32]);
    v8f s0 = vzero(), s1 = vzero();
#pragma unroll
    for (int kc = 0; kc < 4; kc++) s0 = wmma_bf16(qf[kc], kf0[kc], s0);
#pragma unroll
    for (int kc = 0; kc < 4; kc++) s1 = wmma_bf16(qf[kc], kf1[kc], s1);

    // Online softmax: each row's 16 values live across a 16-lane half.
#pragma unroll
    for (int r = 0; r < 8; r++) {
      const float a0 = s0[r] * scale, a1 = s1[r] * scale;
      float mx = fmaxf(a0, a1);
      mx = fmaxf(mx, __shfl_xor(mx, 1));
      mx = fmaxf(mx, __shfl_xor(mx, 2));
      mx = fmaxf(mx, __shfl_xor(mx, 4));
      mx = fmaxf(mx, __shfl_xor(mx, 8));
      const float mnew = fmaxf(mrow[r], mx);
      const float p0 = __expf(a0 - mnew), p1 = __expf(a1 - mnew);
      float sum = p0 + p1;
      sum += __shfl_xor(sum, 1);
      sum += __shfl_xor(sum, 2);
      sum += __shfl_xor(sum, 4);
      sum += __shfl_xor(sum, 8);
      const float alpha = __expf(mrow[r] - mnew);
      lrow[r] = lrow[r] * alpha + sum;
      mrow[r] = mnew;
#pragma unroll
      for (int n = 0; n < 8; n++) oacc[n][r] *= alpha;
      Pl[wave][(rb + r) * 40 + cl]      = f2bf(p0);
      Pl[wave][(rb + r) * 40 + 16 + cl] = f2bf(p1);
    }
    __syncthreads();                    // Pl visible (staging hits other buffer)

    // O += P (16x32) @ V (32x128); V frags preloaded in groups of 4.
    const v16bf pf = ldfrag(&Pl[wave][(lane & 15) * 40 + (lane >> 4) * 8]);
    v16bf vf[4];
#pragma unroll
    for (int n = 0; n < 4; n++)
      vf[n] = ldfrag(&Vt[buf][(n * 16 + (lane & 15)) * 40 + (lane >> 4) * 8]);
#pragma unroll
    for (int n = 0; n < 4; n++) oacc[n] = wmma_bf16(pf, vf[n], oacc[n]);
#pragma unroll
    for (int n = 0; n < 4; n++)
      vf[n] = ldfrag(&Vt[buf][((n + 4) * 16 + (lane & 15)) * 40 + (lane >> 4) * 8]);
#pragma unroll
    for (int n = 0; n < 4; n++) oacc[n + 4] = wmma_bf16(pf, vf[n], oacc[n + 4]);
    buf ^= 1;
  }

  __bf16* ob = o + ((size_t)b * S) * rs + h * HEAD_DIM;
#pragma unroll
  for (int n = 0; n < 8; n++) {
#pragma unroll
    for (int r = 0; r < 8; r++) {
      ob[(size_t)(q0 + rb + r) * rs + n * 16 + cl] = f2bf(oacc[n][r] / lrow[r]);
    }
  }
}

// ---------------------------------------------------------------------------
// Launch: rmsnorm -> QKV gemms -> flash attn -> O proj(+resid) ->
//         rmsnorm -> FFN1+SwiGLU -> FFN2(+resid) -> d_out (fp32)
// ---------------------------------------------------------------------------
extern "C" void kernel_launch(void* const* d_in, const int* in_sizes, int n_in,
                              void* d_out, int out_size, void* d_ws, size_t ws_size,
                              hipStream_t stream) {
  (void)in_sizes; (void)n_in; (void)out_size; (void)ws_size;
  const float* x      = (const float*)d_in[0];
  const float* Wq     = (const float*)d_in[1];
  const float* bq     = (const float*)d_in[2];
  const float* Wk     = (const float*)d_in[3];
  const float* bk     = (const float*)d_in[4];
  const float* Wv     = (const float*)d_in[5];
  const float* bv     = (const float*)d_in[6];
  const float* Wo     = (const float*)d_in[7];
  const float* bo     = (const float*)d_in[8];
  const float* rotary = (const float*)d_in[9];
  const float* n1w    = (const float*)d_in[10];
  const float* n2w    = (const float*)d_in[11];
  const float* W1     = (const float*)d_in[12];
  const float* b1     = (const float*)d_in[13];
  const float* W2     = (const float*)d_in[14];
  const float* b2     = (const float*)d_in[15];
  float* out = (float*)d_out;

  const int B = 2, S = 2048, M = B * S;

  char* ws = (char*)d_ws;
  size_t off = 0;
  auto alloc = [&](size_t bytes) -> void* {
    void* p = ws + off;
    off = (off + bytes + 255) & ~(size_t)255;
    return p;
  };
  __bf16* h    = (__bf16*)alloc((size_t)M * D_MODEL * 2);
  __bf16* qb   = (__bf16*)alloc((size_t)M * D_MODEL * 2);
  __bf16* kb   = (__bf16*)alloc((size_t)M * D_MODEL * 2);
  __bf16* vb   = (__bf16*)alloc((size_t)M * D_MODEL * 2);
  __bf16* ob   = (__bf16*)alloc((size_t)M * D_MODEL * 2);
  float*  xmid = (float*) alloc((size_t)M * D_MODEL * 4);
  __bf16* h2   = (__bf16*)alloc((size_t)M * D_MODEL * 2);
  __bf16* ff   = (__bf16*)alloc((size_t)M * D_FF * 2);

  const dim3 gNorm(M);
  const dim3 gProj(M / 128, D_MODEL / 64);
  const dim3 gF1(M / 128, D_FF / 64);
  const dim3 gAttn(B * N_HEADS, S / 64);

  rmsnorm_kernel<<<gNorm, 256, 0, stream>>>(x, n1w, h, D_MODEL);
  gemm_wmma_kernel<true,  false, false><<<gProj, 256, 0, stream>>>(
      h, Wq, bq, nullptr, rotary, qb, M, D_MODEL, D_MODEL);
  gemm_wmma_kernel<false, false, false><<<gProj, 256, 0, stream>>>(
      h, Wk, bk, nullptr, nullptr, kb, M, D_MODEL, D_MODEL);
  gemm_wmma_kernel<false, false, false><<<gProj, 256, 0, stream>>>(
      h, Wv, bv, nullptr, nullptr, vb, M, D_MODEL, D_MODEL);
  flash_attn_kernel<<<gAttn, 128, 0, stream>>>(qb, kb, vb, ob, S);
  gemm_wmma_kernel<false, true,  false><<<gProj, 256, 0, stream>>>(
      ob, Wo, bo, x, nullptr, xmid, M, D_MODEL, D_MODEL);
  rmsnorm_kernel<<<gNorm, 256, 0, stream>>>(xmid, n2w, h2, D_MODEL);
  gemm_wmma_kernel<false, false, true><<<gF1, 256, 0, stream>>>(
      h2, W1, b1, nullptr, nullptr, ff, M, D_FF, D_MODEL);
  gemm_wmma_kernel<false, true,  false><<<gProj, 256, 0, stream>>>(
      ff, W2, b2, xmid, nullptr, out, M, D_MODEL, D_FF);
}
